// DecoderVDP_50388556316853
// MI455X (gfx1250) — matrix operand order repaired and checked by reading, not compiled, not run on hardware
//
#include <hip/hip_runtime.h>
#include <math.h>

// ---------------------------------------------------------------------------
// Types for CDNA5 WMMA (wave32): bf16 16x16x32 -> f32 accumulate
// ---------------------------------------------------------------------------
typedef __attribute__((ext_vector_type(16))) __bf16 bf16x16;
typedef __attribute__((ext_vector_type(8)))  __bf16 bf16x8;
typedef __attribute__((ext_vector_type(8)))  float  f32x8;

#define NROW 8192   // B*T
#define TT   128
#define HH   8
#define DHH  64
#define KK   512
#define INV_RD  0.0441941738241592f   // 1/sqrt(512)
#define INV_RD2 0.001953125f          // 1/512

__device__ __forceinline__ f32x8 wmma_bf16(bf16x16 a, bf16x16 b, f32x8 c) {
  // D = A(16x32) * B(32x16) + C, f32 accumulate
  return __builtin_amdgcn_wmma_f32_16x16x32_bf16(false, a, false, b, (short)0, c,
                                                 false, false);
}

// A-fragment: 16x32 bf16 from row-major [row][k] LDS, stride ld (elements).
// Per ISA 7.12.2: lanes 0-15 row M=lane, K chunks {0..7,16..23};
//                 lanes 16-31 row M=lane-16, K chunks {8..15,24..31}.
__device__ __forceinline__ bf16x16 fragA(const __bf16* base, int row0, int k0, int ld) {
  int lane = threadIdx.x & 31;
  const __bf16* p = base + (row0 + (lane & 15)) * ld + k0 + ((lane & 16) ? 8 : 0);
  union { bf16x16 f; bf16x8 h[2]; } u;
  u.h[0] = *(const bf16x8*)p;
  u.h[1] = *(const bf16x8*)(p + 16);
  return u.f;
}

// B-fragment: 32x16 bf16, read from B^T stored row-major [n][k].
// lanes 0-15: col N=lane, K {0..7,16..23}; lanes 16-31: col N=lane-16, K {8..15,24..31}.
__device__ __forceinline__ bf16x16 fragB(const __bf16* base, int n0, int k0, int ld) {
  int lane = threadIdx.x & 31;
  const __bf16* p = base + (n0 + (lane & 15)) * ld + k0 + ((lane & 16) ? 8 : 0);
  union { bf16x16 f; bf16x8 h[2]; } u;
  u.h[0] = *(const bf16x8*)p;
  u.h[1] = *(const bf16x8*)(p + 16);
  return u.f;
}

// ---------------------------------------------------------------------------
// Weight prep: transpose W(din x dout) -> WT(NpadxKpad) bf16, plus VT and
// SVT = (V + W^2)^T, zero-padded.
// ---------------------------------------------------------------------------
__global__ void prep_w(const float* __restrict__ W, const float* __restrict__ V,
                       int din, int dout, int Kpad, int Npad,
                       __bf16* __restrict__ WT, __bf16* __restrict__ VT,
                       __bf16* __restrict__ SVT) {
  long i = (long)blockIdx.x * blockDim.x + threadIdx.x;
  if (i >= (long)Npad * Kpad) return;
  int c = (int)(i % Kpad);          // input dim
  int r = (int)(i / Kpad);          // output dim
  float w = 0.f, v = 0.f;
  if (r < dout && c < din) {
    w = W[(long)c * dout + r];
    v = V[(long)c * dout + r];
  }
  WT[i]  = (__bf16)w;
  VT[i]  = (__bf16)v;
  SVT[i] = (__bf16)(v + w * w);
}

// ---------------------------------------------------------------------------
// Workhorse GEMM: C(M x N) = sum_p sqmaybe(Ap) @ Bp  (+ bias), bf16 WMMA.
// A: f32 global (converted to bf16 in LDS, optional elementwise square,
//    K zero-padded to Kpad). B: prepped bf16 B^T (N x Kpad).
// Block = 256 thr = 8 waves, tile 128x64; wave grid 4x2, each wave a 32x32
// (2x2 WMMA-tile) region -> 4 WMMAs per 4 fragment loads per product.
// ---------------------------------------------------------------------------
__global__ __launch_bounds__(256)
void gemm_vdp(float* __restrict__ C, int ldc,
              const float* __restrict__ A1, int lda1, int sq1,
              const __bf16* __restrict__ B1,
              const float* __restrict__ A2, int lda2, int sq2,
              const __bf16* __restrict__ B2,
              int Kd, int Kpad, const float* __restrict__ bias, int nbias) {
  __shared__ alignas(16) __bf16 As[2][128][40];
  __shared__ alignas(16) __bf16 Bs[2][64][40];
  const int tn0 = blockIdx.x * 64;
  const int tm0 = blockIdx.y * 128;
  const int tid = threadIdx.x;
  const int w   = tid >> 5;
  const int wr  = w & 3;            // wave row group (32 rows)
  const int wc  = w >> 2;           // wave col group (32 cols)
  const int nprod = (A2 != nullptr) ? 2 : 1;
  const int alr = tid >> 1;         // 0..127: A row to load
  const int alc = (tid & 1) << 4;   // 0/16: A k chunk (16 elems)
  const int blr = tid >> 2;         // 0..63: B row to load
  const int blc = (tid & 3) << 3;   // 0,8,16,24
  f32x8 acc[2][2];
#pragma unroll
  for (int i = 0; i < 2; ++i)
#pragma unroll
    for (int j = 0; j < 2; ++j) acc[i][j] = (f32x8){0, 0, 0, 0, 0, 0, 0, 0};

  for (int k0 = 0; k0 < Kpad; k0 += 32) {
    for (int pr = 0; pr < nprod; ++pr) {
      const float* A  = pr ? A2 : A1;
      const int lda   = pr ? lda2 : lda1;
      const int sq    = pr ? sq2 : sq1;
      const __bf16* B = pr ? B2 : B1;
      const float* ap = A + (long)(tm0 + alr) * lda + (k0 + alc);
      if (k0 + 32 < Kpad)                 // gfx1250 global_prefetch_b8
        __builtin_prefetch(ap + 32, 0, 0);
#pragma unroll
      for (int e = 0; e < 16; ++e) {
        int kg = k0 + alc + e;
        float v = (kg < Kd) ? ap[e] : 0.0f;
        if (sq) v *= v;
        As[pr][alr][alc + e] = (__bf16)v;
      }
      const __bf16* bp = B + (long)(tn0 + blr) * Kpad + (k0 + blc);
#pragma unroll
      for (int e = 0; e < 8; ++e) Bs[pr][blr][blc + e] = bp[e];
    }
    __syncthreads();
#pragma unroll
    for (int pr = 0; pr < 2; ++pr) {
      if (pr < nprod) {
        bf16x16 a0 = fragA(&As[pr][0][0], wr * 32,      0, 40);
        bf16x16 a1 = fragA(&As[pr][0][0], wr * 32 + 16, 0, 40);
        bf16x16 b0 = fragB(&Bs[pr][0][0], wc * 32,      0, 40);
        bf16x16 b1 = fragB(&Bs[pr][0][0], wc * 32 + 16, 0, 40);
        acc[0][0] = wmma_bf16(a0, b0, acc[0][0]);
        acc[0][1] = wmma_bf16(a0, b1, acc[0][1]);
        acc[1][0] = wmma_bf16(a1, b0, acc[1][0]);
        acc[1][1] = wmma_bf16(a1, b1, acc[1][1]);
      }
    }
    __syncthreads();
  }
  // Epilogue: C/D layout — VGPR r: lanes0-15 M=r, lanes16-31 M=8+r, N=lane%16
  const int lane = tid & 31;
  const int col  = lane & 15;
  const int ro   = (lane >> 4) << 3;
#pragma unroll
  for (int i = 0; i < 2; ++i) {
#pragma unroll
    for (int j = 0; j < 2; ++j) {
      int n = tn0 + wc * 32 + j * 16 + col;
      float bb = (bias != nullptr && n < nbias) ? bias[n] : 0.0f;
#pragma unroll
      for (int r = 0; r < 8; ++r)
        C[(long)(tm0 + wr * 32 + i * 16 + ro + r) * ldc + n] = acc[i][j][r] + bb;
    }
  }
}

// ---------------------------------------------------------------------------
// Fused masked VDP attention for one (b,h): S/VS WMMA -> register softmax_vdp
// -> P/VP/P^2 bf16 in LDS -> O/VO WMMA -> residual epilogue (X += O, VX = VO).
// ---------------------------------------------------------------------------
#define LQ 72
#define LP 136
#define ATTN_LDS ((6 * 128 * LQ + 3 * 64 * LP) * 2)   // 162816 bytes

__global__ __launch_bounds__(256)
void attn_vdp(const float* __restrict__ Qm, const float* __restrict__ Qv,
              long qBS, long qHS, int qRS,
              const float* __restrict__ Km, const float* __restrict__ Kv,
              long kBS, long kHS, int kRS,
              const float* __restrict__ Vm, const float* __restrict__ Vv,
              long vBS, long vHS, int vRS,
              float* __restrict__ X, float* __restrict__ VX) {
  extern __shared__ char smem[];
  __bf16* sqm  = (__bf16*)smem;          // [128][72] q mean
  __bf16* sqv  = sqm  + 128 * LQ;        // vq
  __bf16* sq2m = sqv  + 128 * LQ;        // q^2
  __bf16* skm  = sq2m + 128 * LQ;        // k mean
  __bf16* skv2 = skm  + 128 * LQ;        // vk + k^2
  __bf16* skv  = skv2 + 128 * LQ;        // vk
  // P region overlays q/k region after barrier (3*128*136 <= 6*128*72)
  __bf16* pm = (__bf16*)smem;            // [128][136]
  __bf16* pv = pm + 128 * LP;
  __bf16* p2 = pv + 128 * LP;
  __bf16* vtm  = (__bf16*)(smem + 6 * 128 * LQ * 2);  // V^T [64][136] mean
  __bf16* vtv2 = vtm  + 64 * LP;                      // (vv + v^2)^T
  __bf16* vtv  = vtv2 + 64 * LP;                      // vv^T

  const int bh = blockIdx.x;
  const int bI = bh / HH, hI = bh % HH;
  const int tid = threadIdx.x;
  const float* qm_g = Qm + (long)bI * qBS + (long)hI * qHS;
  const float* qv_g = Qv + (long)bI * qBS + (long)hI * qHS;
  const float* km_g = Km + (long)bI * kBS + (long)hI * kHS;
  const float* kv_g = Kv + (long)bI * kBS + (long)hI * kHS;
  const float* vm_g = Vm + (long)bI * vBS + (long)hI * vHS;
  const float* vv_g = Vv + (long)bI * vBS + (long)hI * vHS;

  for (int i = tid; i < 128 * 64; i += 256) {
    int r = i >> 6, c = i & 63;
    float q0 = qm_g[(long)r * qRS + c], q1 = qv_g[(long)r * qRS + c];
    sqm[r * LQ + c]  = (__bf16)q0;
    sqv[r * LQ + c]  = (__bf16)q1;
    sq2m[r * LQ + c] = (__bf16)(q0 * q0);
    float k0 = km_g[(long)r * kRS + c], k1 = kv_g[(long)r * kRS + c];
    skm[r * LQ + c]  = (__bf16)k0;
    skv[r * LQ + c]  = (__bf16)k1;
    skv2[r * LQ + c] = (__bf16)(k1 + k0 * k0);
    float v0 = vm_g[(long)r * vRS + c], v1 = vv_g[(long)r * vRS + c];
    vtm[c * LP + r]  = (__bf16)v0;     // transposed for B-fragments
    vtv[c * LP + r]  = (__bf16)v1;
    vtv2[c * LP + r] = (__bf16)(v1 + v0 * v0);
  }
  __syncthreads();

  const int w  = tid >> 5;
  const int m0 = w * 16;               // 16-row strip per wave
  const int lane = tid & 31;
  const int col  = lane & 15;
  const int ro   = (lane >> 4) << 3;

  // --- GEMM1: S = q @ k^T ; VS = vq @ (vk+k^2)^T + q^2 @ vk^T ---
  f32x8 Sm[8], Sv[8];
#pragma unroll
  for (int j = 0; j < 8; ++j) {
    Sm[j] = (f32x8){0, 0, 0, 0, 0, 0, 0, 0};
    Sv[j] = (f32x8){0, 0, 0, 0, 0, 0, 0, 0};
  }
#pragma unroll
  for (int kc = 0; kc < 64; kc += 32) {
    bf16x16 am = fragA(sqm,  m0, kc, LQ);
    bf16x16 av = fragA(sqv,  m0, kc, LQ);
    bf16x16 a2 = fragA(sq2m, m0, kc, LQ);
#pragma unroll
    for (int j = 0; j < 8; ++j) {
      bf16x16 bm  = fragB(skm,  j * 16, kc, LQ);
      bf16x16 bv2 = fragB(skv2, j * 16, kc, LQ);
      bf16x16 bv  = fragB(skv,  j * 16, kc, LQ);
      Sm[j] = wmma_bf16(am, bm,  Sm[j]);
      Sv[j] = wmma_bf16(av, bv2, Sv[j]);
      Sv[j] = wmma_bf16(a2, bv,  Sv[j]);
    }
  }
  __syncthreads();   // all q/k reads done before P overlay writes

  // --- scale + causal mask + softmax_vdp in registers ---
#pragma unroll
  for (int r = 0; r < 8; ++r) {
    int m = m0 + ro + r;               // global row, same for all 16 lanes/half
    float sloc[8];
    float mx = -INFINITY;
#pragma unroll
    for (int j = 0; j < 8; ++j) {
      int n = j * 16 + col;
      float s = Sm[j][r] * INV_RD;
      if (n > m) s = -INFINITY;
      sloc[j] = s;
      mx = fmaxf(mx, s);
    }
#pragma unroll
    for (int d = 8; d >= 1; d >>= 1) mx = fmaxf(mx, __shfl_xor(mx, d));
    float sum = 0.0f;
#pragma unroll
    for (int j = 0; j < 8; ++j) { float p = __expf(sloc[j] - mx); sloc[j] = p; sum += p; }
#pragma unroll
    for (int d = 8; d >= 1; d >>= 1) sum += __shfl_xor(sum, d);
    float inv = 1.0f / sum;
#pragma unroll
    for (int j = 0; j < 8; ++j) {
      int n = j * 16 + col;
      float p  = sloc[j] * inv;
      float vs = Sv[j][r] * INV_RD2;
      float pq = p * (1.0f - p);
      pm[m * LP + n] = (__bf16)p;
      pv[m * LP + n] = (__bf16)(pq * pq * vs);
      p2[m * LP + n] = (__bf16)(p * p);
    }
  }
  __syncthreads();

  // --- GEMM2: O = P @ V ; VO = VP @ (vv+v^2) + P^2 @ vv ---
  f32x8 Om[4], Ov[4];
#pragma unroll
  for (int j = 0; j < 4; ++j) {
    Om[j] = (f32x8){0, 0, 0, 0, 0, 0, 0, 0};
    Ov[j] = (f32x8){0, 0, 0, 0, 0, 0, 0, 0};
  }
#pragma unroll
  for (int kc = 0; kc < 128; kc += 32) {
    bf16x16 am = fragA(pm, m0, kc, LP);
    bf16x16 av = fragA(pv, m0, kc, LP);
    bf16x16 a2 = fragA(p2, m0, kc, LP);
#pragma unroll
    for (int jn = 0; jn < 4; ++jn) {
      bf16x16 bm  = fragB(vtm,  jn * 16, kc, LP);
      bf16x16 bv2 = fragB(vtv2, jn * 16, kc, LP);
      bf16x16 bv  = fragB(vtv,  jn * 16, kc, LP);
      Om[jn] = wmma_bf16(am, bm,  Om[jn]);
      Ov[jn] = wmma_bf16(av, bv2, Ov[jn]);
      Ov[jn] = wmma_bf16(a2, bv,  Ov[jn]);
    }
  }
  // epilogue: X += merge(O) (residual on mean), VX = merge(VO)
#pragma unroll
  for (int jn = 0; jn < 4; ++jn) {
#pragma unroll
    for (int r = 0; r < 8; ++r) {
      int m = m0 + ro + r;
      int n = jn * 16 + col;
      long off = ((long)(bI * TT + m)) * KK + hI * DHH + n;
      X[off] += Om[jn][r];
      VX[off] = Ov[jn][r];
    }
  }
}

// ---------------------------------------------------------------------------
// Elementwise / normalization kernels
// ---------------------------------------------------------------------------
__global__ __launch_bounds__(256)
void ln_vdp(float* __restrict__ X, float* __restrict__ VX,
            const float* __restrict__ g, const float* __restrict__ b) {
  int w = threadIdx.x >> 5, lane = threadIdx.x & 31;
  long row = (long)blockIdx.x * 8 + w;
  float* x  = X  + row * KK;
  float* vx = VX + row * KK;
  float s = 0.f, s2 = 0.f;
  for (int i = lane; i < KK; i += 32) { float v = x[i]; s += v; s2 += v * v; }
#pragma unroll
  for (int d = 16; d >= 1; d >>= 1) { s += __shfl_xor(s, d); s2 += __shfl_xor(s2, d); }
  float mean = s * (1.0f / KK);
  float var  = s2 * (1.0f / KK) - mean * mean;
  float ieps = 1.0f / (var + 1e-5f);
  float rst  = sqrtf(ieps);
  for (int i = lane; i < KK; i += 32) {
    float gg = g[i];
    x[i]  = gg * (x[i] - mean) * rst + b[i];
    vx[i] = vx[i] * gg * gg * ieps;
  }
}

__global__ void relu_pair(float* __restrict__ m, float* __restrict__ v, long n) {
  long i = (long)blockIdx.x * blockDim.x + threadIdx.x;
  if (i >= n) return;
  float mm = m[i];
  if (!(mm > 0.0f)) { m[i] = 0.0f; v[i] = 0.0f; }
}

__global__ void ffn_add_relu(float* __restrict__ X, const float* __restrict__ F, long n) {
  long i = (long)blockIdx.x * blockDim.x + threadIdx.x;
  if (i >= n) return;
  X[i] += fmaxf(F[i], 0.0f);   // residual adds relu(f); variance kept pre-FFN
}

__global__ void posenc_add(float* __restrict__ X) {
  long i = (long)blockIdx.x * blockDim.x + threadIdx.x;
  if (i >= (long)NROW * KK) return;
  int c = (int)(i & (KK - 1));
  long row = i >> 9;
  int t = (int)(row & (TT - 1));
  float fi  = (float)(c >> 1);
  float ang = (float)t * __powf(10000.0f, -(2.0f * fi) / (float)KK);
  X[i] += (c & 1) ? __cosf(ang) : __sinf(ang);
}

__global__ void head_k(const float* __restrict__ M, const float* __restrict__ V,
                       const float* __restrict__ cut, float* __restrict__ out) {
  const long total = (long)NROW * 500;
  long i = (long)blockIdx.x * blockDim.x + threadIdx.x;
  if (i >= total) return;
  long row = i / 500;
  int d = (int)(i - row * 500);
  float x  = M[row * 512 + d];
  float vx = V[row * 512 + d];
  float b0 = cut[0];
  float b1 = b0 + cut[1] * cut[1];
  float b2 = b1 + cut[2] * cut[2];
  float b3 = b2 + cut[3] * cut[3];
  float s0 = 1.0f / (1.0f + __expf(x - b0));
  float s1 = 1.0f / (1.0f + __expf(x - b1));
  float s2 = 1.0f / (1.0f + __expf(x - b2));
  float s3 = 1.0f / (1.0f + __expf(x - b3));
  long o = i * 5;
  out[o + 0] = s0;
  out[o + 1] = s1 - s0;
  out[o + 2] = s2 - s1;
  out[o + 3] = s3 - s2;
  out[o + 4] = 1.0f - s3;
  float t0 = s0 * (1.f - s0), t1 = s1 * (1.f - s1);
  float t2 = s2 * (1.f - s2), t3 = s3 * (1.f - s3);
  float v0 = t0 * t0 * vx, v1 = t1 * t1 * vx, v2 = t2 * t2 * vx, v3 = t3 * t3 * vx;
  long p = total * 5 + o;
  out[p + 0] = v0;
  out[p + 1] = v1 + v0;
  out[p + 2] = v2 + v1;
  out[p + 3] = v3 + v2;
  out[p + 4] = v3;
}

// ---------------------------------------------------------------------------
// Host orchestration
// ---------------------------------------------------------------------------
extern "C" void kernel_launch(void* const* d_in, const int* in_sizes, int n_in,
                              void* d_out, int out_size, void* d_ws, size_t ws_size,
                              hipStream_t stream) {
  (void)in_sizes; (void)n_in; (void)out_size; (void)ws_size;
  // ---- map inputs (setup_inputs() dict insertion order, recursively) ----
  int pi = 0;
  auto nextf = [&]() { return (const float*)d_in[pi++]; };
  const float* x_in  = nextf();
  const float* k_in  = nextf();
  const float* vk_in = nextf();
  const float* v_in  = nextf();
  const float* vv_in = nextf();
  struct Lin { const float *W, *V, *b, *vb; };
  auto lin = [&](bool bias) {
    Lin L; L.W = nextf(); L.V = nextf();
    L.b = bias ? nextf() : nullptr; L.vb = bias ? nextf() : nullptr; return L;
  };
  struct Nrm { const float *g, *b; };
  auto nrm = [&]() { Nrm N; N.g = nextf(); N.b = nextf(); return N; };
  Lin emb0 = lin(true), emb1 = lin(true);
  Lin sqL[4], skL[4], svL[4], cqL[4], fcL[4];
  Nrm n1L[4], n2L[4], n3L[4];
  for (int i = 0; i < 4; ++i) {
    sqL[i] = lin(false); skL[i] = lin(false); svL[i] = lin(false);
    n2L[i] = nrm(); cqL[i] = lin(false); n3L[i] = nrm(); fcL[i] = lin(true);
    if (i != 0) n1L[i] = nrm();   // n1 appended last in the layer dict
  }
  Lin fco = lin(true);
  const float* cut = nextf();

  // ---- workspace bump allocator ----
  size_t off = 0;
  auto alloc = [&](size_t bytes) {
    void* p = (char*)d_ws + off;
    off += (bytes + 255) & ~(size_t)255;
    return p;
  };
  const size_t actB = (size_t)NROW * 512 * sizeof(float);
  float* X  = (float*)alloc(actB);
  float* VX = (float*)alloc(actB);
  float* T1 = (float*)alloc(actB);
  float* T2 = (float*)alloc(actB);
  float* T3 = (float*)alloc(actB);
  float* T4 = (float*)alloc(actB);
  float* T5 = (float*)alloc(actB);
  float* T6 = (float*)alloc(actB);

  struct Trio { __bf16 *WT, *VT, *SVT; };
  auto prep = [&](const Lin& L, int din, int dout, int Kpad, int Npad) {
    Trio t;
    size_t mb = (size_t)Npad * Kpad * sizeof(__bf16);
    t.WT = (__bf16*)alloc(mb); t.VT = (__bf16*)alloc(mb); t.SVT = (__bf16*)alloc(mb);
    long n = (long)Npad * Kpad;
    prep_w<<<dim3((unsigned)((n + 255) / 256)), 256, 0, stream>>>(
        L.W, L.V, din, dout, Kpad, Npad, t.WT, t.VT, t.SVT);
    return t;
  };
  Trio tE0 = prep(emb0, 500, 256, 512, 256);
  Trio tE1 = prep(emb1, 256, 512, 256, 512);
  Trio tSq[4], tSk[4], tSv[4], tCq[4], tFc[4];
  for (int i = 0; i < 4; ++i) {
    tSq[i] = prep(sqL[i], 512, 512, 512, 512);
    tSk[i] = prep(skL[i], 512, 512, 512, 512);
    tSv[i] = prep(svL[i], 512, 512, 512, 512);
    tCq[i] = prep(cqL[i], 512, 512, 512, 512);
    tFc[i] = prep(fcL[i], 512, 512, 512, 512);
  }
  Trio tOut = prep(fco, 512, 500, 512, 512);

  auto gemm = [&](float* C, int N,
                  const float* A1, int lda1, bool s1, const __bf16* B1,
                  const float* A2, int lda2, bool s2, const __bf16* B2,
                  int Kd, int Kpad, const float* bias, int nbias) {
    dim3 grid(N / 64, NROW / 128);
    gemm_vdp<<<grid, 256, 0, stream>>>(C, N, A1, lda1, s1 ? 1 : 0, B1,
                                       A2, lda2, s2 ? 1 : 0, B2,
                                       Kd, Kpad, bias, nbias);
  };
  auto ln = [&](const Nrm& N) {
    ln_vdp<<<NROW / 8, 256, 0, stream>>>(X, VX, N.g, N.b);
  };

  (void)hipFuncSetAttribute((const void*)attn_vdp,
                            hipFuncAttributeMaxDynamicSharedMemorySize, ATTN_LDS);

  // ---- embedding MLP ----
  gemm(T1, 256, x_in, 500, false, tE0.WT, nullptr, 0, false, nullptr, 500, 512, emb0.b, 256);
  gemm(T2, 256, x_in, 500, true,  tE0.VT, nullptr, 0, false, nullptr, 500, 512, emb0.vb, 256);
  relu_pair<<<(unsigned)((long)NROW * 256 / 256), 256, 0, stream>>>(T1, T2, (long)NROW * 256);
  gemm(X,  512, T1, 256, false, tE1.WT,  nullptr, 0, false, nullptr, 256, 256, emb1.b, 512);
  gemm(VX, 512, T2, 256, false, tE1.SVT, T1, 256, true, tE1.VT,     256, 256, emb1.vb, 512);
  posenc_add<<<(unsigned)((long)NROW * 512 / 256), 256, 0, stream>>>(X);

  const long pBS = (long)TT * KK; const long pHS = DHH; const int pRS = KK;        // proj layout
  const long eBS = (long)HH * TT * DHH; const long eHS = (long)TT * DHH; const int eRS = DHH;  // encoder layout

  for (int i = 0; i < 4; ++i) {
    if (i != 0) ln(n1L[i]);
    // self-attn projections (no bias)
    gemm(T1, 512, X,  512, false, tSq[i].WT,  nullptr, 0, false, nullptr, 512, 512, nullptr, 0);
    gemm(T2, 512, VX, 512, false, tSq[i].SVT, X, 512, true, tSq[i].VT,   512, 512, nullptr, 0);
    gemm(T3, 512, X,  512, false, tSk[i].WT,  nullptr, 0, false, nullptr, 512, 512, nullptr, 0);
    gemm(T4, 512, VX, 512, false, tSk[i].SVT, X, 512, true, tSk[i].VT,   512, 512, nullptr, 0);
    gemm(T5, 512, X,  512, false, tSv[i].WT,  nullptr, 0, false, nullptr, 512, 512, nullptr, 0);
    gemm(T6, 512, VX, 512, false, tSv[i].SVT, X, 512, true, tSv[i].VT,   512, 512, nullptr, 0);
    attn_vdp<<<64 * HH, 256, ATTN_LDS, stream>>>(T1, T2, pBS, pHS, pRS,
                                                 T3, T4, pBS, pHS, pRS,
                                                 T5, T6, pBS, pHS, pRS, X, VX);
    ln(n2L[i]);
    // cross-attn (encoder k/v from inputs)
    gemm(T1, 512, X,  512, false, tCq[i].WT,  nullptr, 0, false, nullptr, 512, 512, nullptr, 0);
    gemm(T2, 512, VX, 512, false, tCq[i].SVT, X, 512, true, tCq[i].VT,   512, 512, nullptr, 0);
    attn_vdp<<<64 * HH, 256, ATTN_LDS, stream>>>(T1, T2, pBS, pHS, pRS,
                                                 k_in, vk_in, eBS, eHS, eRS,
                                                 v_in, vv_in, eBS, eHS, eRS, X, VX);
    ln(n3L[i]);
    // FFN: variance path is discarded by the reference (vx keeps pre-FFN value)
    gemm(T1, 512, X, 512, false, tFc[i].WT, nullptr, 0, false, nullptr, 512, 512, fcL[i].b, 512);
    ffn_add_relu<<<(unsigned)((long)NROW * 512 / 256), 256, 0, stream>>>(X, T1, (long)NROW * 512);
  }

  // ---- output projection (padded to 512 cols) + ordinal head ----
  gemm(T1, 512, X,  512, false, tOut.WT,  nullptr, 0, false, nullptr, 512, 512, fco.b, 500);
  gemm(T2, 512, VX, 512, false, tOut.SVT, X, 512, true, tOut.VT,     512, 512, fco.vb, 500);
  head_k<<<(unsigned)(((long)NROW * 500 + 255) / 256), 256, 0, stream>>>(
      T1, T2, cut, (float*)d_out);
}
